// SimpleModalityUntiedAttention_48687749268002
// MI455X (gfx1250) — compile-verified, matrix operand order checked by reading, not run on hardware
//
#include <hip/hip_runtime.h>

typedef __attribute__((ext_vector_type(16))) __bf16 v16bf;
typedef __attribute__((ext_vector_type(8)))  __bf16 bf16x8;
typedef __attribute__((ext_vector_type(4)))  __bf16 bf16x4;
typedef __attribute__((ext_vector_type(8)))  float  v8f;

#define B_DIM  2
#define S_DIM  2048
#define D_DIM  1024
#define E_DIM  1024
#define H_DIM  16
#define HD_DIM 64
#define ROWS   (B_DIM * S_DIM)   // 4096
#define EPS_F  1e-5f

// ---------------------------------------------------------------------------
// Fragment loader: 16 bf16 per lane = two contiguous 16B loads.
// Lanes 0-15 hold K {base..+7, base+16..+23}; lanes 16-31 {base+8..+15,+24..+31};
// caller adds half*8 to the pointer.
// ---------------------------------------------------------------------------
__device__ inline v16bf load_frag(const __bf16* p) {
  bf16x8 lo = *(const bf16x8*)p;
  bf16x8 hi = *(const bf16x8*)(p + 16);
  v16bf a;
#pragma unroll
  for (int j = 0; j < 8; ++j) { a[j] = lo[j]; a[j + 8] = hi[j]; }
  return a;
}

// Build a bf16 fragment from fp32 probabilities in LDS (32B-aligned segments).
__device__ inline v16bf frag_from_lds(const float* pr) {
  const float4* p4 = (const float4*)pr;
  float4 a0 = p4[0], a1 = p4[1];   // pr[0..7]
  float4 b0 = p4[4], b1 = p4[5];   // pr[16..23]
  v16bf f;
  f[0] = (__bf16)a0.x; f[1] = (__bf16)a0.y; f[2]  = (__bf16)a0.z; f[3]  = (__bf16)a0.w;
  f[4] = (__bf16)a1.x; f[5] = (__bf16)a1.y; f[6]  = (__bf16)a1.z; f[7]  = (__bf16)a1.w;
  f[8] = (__bf16)b0.x; f[9] = (__bf16)b0.y; f[10] = (__bf16)b0.z; f[11] = (__bf16)b0.w;
  f[12] = (__bf16)b1.x; f[13] = (__bf16)b1.y; f[14] = (__bf16)b1.z; f[15] = (__bf16)b1.w;
  return f;
}

// ---------------------------------------------------------------------------
// Vectorized fp32 -> bf16 convert (4 elems / thread)
// ---------------------------------------------------------------------------
__global__ void cvt_f32_bf16(const float* __restrict__ in, __bf16* __restrict__ o, int n4) {
  int i = blockIdx.x * 256 + threadIdx.x;
  if (i < n4) {
    float4 v = ((const float4*)in)[i];
    bf16x4 r;
    r[0] = (__bf16)v.x; r[1] = (__bf16)v.y; r[2] = (__bf16)v.z; r[3] = (__bf16)v.w;
    ((bf16x4*)o)[i] = r;
  }
}

// W: [M][K][N] fp32 -> WT: [M][N][K] bf16 (so B-fragments read contiguous K)
__global__ void transpose_cvt(const float* __restrict__ W, __bf16* __restrict__ WT,
                              int K, int N) {
  int i = blockIdx.x * 256 + threadIdx.x;
  int total = 2 * K * N;
  if (i >= total) return;
  int m   = i / (K * N);
  int rem = i - m * (K * N);
  int n   = rem / K;
  int k   = rem - n * K;
  WT[i] = (__bf16)W[(size_t)(m * K + k) * N + n];
}

// ---------------------------------------------------------------------------
// Routed GEMM:  C[row, n] = sum_k A[row,k] * Bt[m][n,k]
// Block = 256 thr (8 waves); tile = 128 rows x 64 cols; wave owns 16x64
// (4 independent WMMA accumulator chains). grid = (ROWS/128, N/64, M).
// Epilogue stores only rows with mod_ids[row]==m.
// MODE 0: fused per-head(64-col) RMSNorm * normw -> bf16 row-major   (q, k)
// MODE 1: bf16 stored transposed as [B][E][S]                        (v)
// MODE 2: fp32 row-major raw                                         (o_raw)
// ---------------------------------------------------------------------------
template <int MODE>
__global__ __launch_bounds__(256) void gemm_routed(
    const __bf16* __restrict__ A,    // [ROWS][1024] bf16 row-major
    const __bf16* __restrict__ Bt,   // [M][N=1024][K=1024] bf16
    void* __restrict__ outp,
    const int* __restrict__ mod_ids,
    const float* __restrict__ normw) // [M][64] for MODE 0
{
  const int wave = threadIdx.x >> 5;
  const int lane = threadIdx.x & 31;
  const int l16  = lane & 15;
  const int half = lane >> 4;
  const int m    = blockIdx.z;
  const int tileRow = blockIdx.x * 128 + wave * 16;
  const int colBase = blockIdx.y * 64;

  const __bf16* Bm   = Bt + (size_t)m * E_DIM * D_DIM;
  const int     koff = half * 8;

  const __bf16* aptr = A + (size_t)(tileRow + l16) * D_DIM + koff;
  const __bf16* bptr[4];
#pragma unroll
  for (int a = 0; a < 4; ++a)
    bptr[a] = Bm + (size_t)(colBase + 16 * a + l16) * D_DIM + koff;

  v8f acc[4] = {v8f{}, v8f{}, v8f{}, v8f{}};

  for (int kb = 0; kb < D_DIM; kb += 32) {
    __builtin_prefetch(aptr + 128, 0, 0);       // 4 k-steps ahead
    __builtin_prefetch(bptr[0] + 128, 0, 0);
    __builtin_prefetch(bptr[2] + 128, 0, 0);
    v16bf af = load_frag(aptr);
    aptr += 32;
#pragma unroll
    for (int a = 0; a < 4; ++a) {
      v16bf bf = load_frag(bptr[a]);
      bptr[a] += 32;
      acc[a] = __builtin_amdgcn_wmma_f32_16x16x32_bf16(
          false, af, false, bf, (short)0, acc[a], false, false);
    }
  }

  // ---- epilogue ----
  float ss[8];
  float wn[4];
  if (MODE == 0) {
    // per-row sum of squares over the 64-col head tile (16 lanes share a row)
#pragma unroll
    for (int i = 0; i < 8; ++i) {
      float t = 0.f;
#pragma unroll
      for (int a = 0; a < 4; ++a) { float v = acc[a][i]; t += v * v; }
      ss[i] = t;
    }
#pragma unroll
    for (int mk = 1; mk < 16; mk <<= 1) {
#pragma unroll
      for (int i = 0; i < 8; ++i) ss[i] += __shfl_xor(ss[i], mk, 32);
    }
#pragma unroll
    for (int a = 0; a < 4; ++a) wn[a] = normw[m * HD_DIM + 16 * a + l16];
  }

#pragma unroll
  for (int i = 0; i < 8; ++i) {
    int row = tileRow + i + half * 8;
    if (mod_ids[row] != m) continue;
    if (MODE == 0) {
      float sc = rsqrtf(ss[i] * (1.0f / HD_DIM) + EPS_F);
      __bf16* o = (__bf16*)outp;
#pragma unroll
      for (int a = 0; a < 4; ++a)
        o[(size_t)row * E_DIM + colBase + 16 * a + l16] =
            (__bf16)(acc[a][i] * sc * wn[a]);
    } else if (MODE == 1) {
      __bf16* o = (__bf16*)outp;
      int bb = row >> 11, s = row & (S_DIM - 1);
#pragma unroll
      for (int a = 0; a < 4; ++a)
        o[((size_t)bb * E_DIM + colBase + 16 * a + l16) * S_DIM + s] =
            (__bf16)acc[a][i];
    } else {
      float* o = (float*)outp;
#pragma unroll
      for (int a = 0; a < 4; ++a)
        o[(size_t)row * D_DIM + colBase + 16 * a + l16] = acc[a][i];
    }
  }
}

// ---------------------------------------------------------------------------
// Attention: one block per (b, 16-row q tile). Loops over 16 heads.
// Dynamic LDS: scores[16][2048] + headMean[16][2048] fp32 (256 KB).
// scores via WMMA (q x k^T, 2 independent chains), softmax fp32 (float4),
// mean accumulated in LDS, ctx = probs x V with 8 waves / 4-way acc chains.
// ---------------------------------------------------------------------------
__global__ __launch_bounds__(256) void attn_kernel(
    const __bf16* __restrict__ qb, const __bf16* __restrict__ kb,
    const __bf16* __restrict__ vT,   // [B][E][S]
    __bf16* __restrict__ ctx,        // [ROWS][E]
    const float* __restrict__ mask,  // [S][S] additive
    float* __restrict__ attn_out)    // [B][S][S]
{
  extern __shared__ float lds[];
  float* scoreL = lds;                 // 16*2048
  float* meanL  = lds + 16 * S_DIM;    // 16*2048
  __shared__ float red[16 * 16];
  __shared__ float rowred[16];
  __shared__ float ctxred[4 * 32 * 8]; // 4 ntiles x 32 lanes x 8 rows

  const int tid  = threadIdx.x;
  const int wave = tid >> 5, lane = tid & 31, l16 = lane & 15, half = lane >> 4;
  const int b  = blockIdx.x >> 7;        // / 128
  const int qt = blockIdx.x & 127;
  const int q0 = b * S_DIM + qt * 16;    // first global q row

  {
    float4 z = make_float4(0.f, 0.f, 0.f, 0.f);
    float4* m4 = (float4*)meanL;
    for (int i = tid; i < 16 * S_DIM / 4; i += 256) m4[i] = z;
  }
  __syncthreads();

  const int row = tid >> 4;   // softmax: 16 threads per row
  const int seg = tid & 15;

  for (int h = 0; h < H_DIM; ++h) {
    const int ecol = h * HD_DIM;

    // ---- scores: 8 waves, 2 key tiles (independent WMMA chains) per iter ----
    const __bf16* qrow = qb + (size_t)(q0 + l16) * E_DIM + ecol + half * 8;
    v16bf qf0 = load_frag(qrow);
    v16bf qf1 = load_frag(qrow + 32);
    for (int kt = wave; kt < S_DIM / 16; kt += 16) {
      const int kt2 = kt + 8;
      const __bf16* kr0 = kb + (size_t)(b * S_DIM + kt  * 16 + l16) * E_DIM + ecol + half * 8;
      const __bf16* kr1 = kb + (size_t)(b * S_DIM + kt2 * 16 + l16) * E_DIM + ecol + half * 8;
      __builtin_prefetch(kb + (size_t)(b * S_DIM + (kt + 16) * 16 + l16) * E_DIM + ecol, 0, 0);
      v16bf b00 = load_frag(kr0);
      v16bf b01 = load_frag(kr0 + 32);
      v16bf b10 = load_frag(kr1);
      v16bf b11 = load_frag(kr1 + 32);
      v8f a0 = {}, a1 = {};
      a0 = __builtin_amdgcn_wmma_f32_16x16x32_bf16(false, qf0, false, b00, (short)0, a0, false, false);
      a1 = __builtin_amdgcn_wmma_f32_16x16x32_bf16(false, qf0, false, b10, (short)0, a1, false, false);
      a0 = __builtin_amdgcn_wmma_f32_16x16x32_bf16(false, qf1, false, b01, (short)0, a0, false, false);
      a1 = __builtin_amdgcn_wmma_f32_16x16x32_bf16(false, qf1, false, b11, (short)0, a1, false, false);
      const int col0 = kt * 16 + l16, col1 = kt2 * 16 + l16;
#pragma unroll
      for (int i = 0; i < 8; ++i) {
        int r = i + half * 8;
        const float* mrow = mask + (size_t)(qt * 16 + r) * S_DIM;
        scoreL[r * S_DIM + col0] = a0[i] * 0.125f + mrow[col0];
        scoreL[r * S_DIM + col1] = a1[i] * 0.125f + mrow[col1];
      }
    }
    __syncthreads();

    // ---- softmax over each 2048-wide row (float4 LDS traffic) ----
    float4* sr = (float4*)(scoreL + row * S_DIM + seg * 128);
    float mx = -3.0e38f;
    for (int j = 0; j < 32; ++j) {
      float4 v = sr[j];
      mx = fmaxf(mx, fmaxf(fmaxf(v.x, v.y), fmaxf(v.z, v.w)));
    }
    red[row * 16 + seg] = mx;
    __syncthreads();
    if (seg == 0) {
      float t = red[row * 16];
      for (int j = 1; j < 16; ++j) t = fmaxf(t, red[row * 16 + j]);
      rowred[row] = t;
    }
    __syncthreads();
    float rm  = rowred[row];
    float sum = 0.f;
    for (int j = 0; j < 32; ++j) {
      float4 v = sr[j];
      v.x = __expf(v.x - rm); v.y = __expf(v.y - rm);
      v.z = __expf(v.z - rm); v.w = __expf(v.w - rm);
      sr[j] = v;
      sum += v.x + v.y + v.z + v.w;
    }
    red[row * 16 + seg] = sum;
    __syncthreads();
    if (seg == 0) {
      float t = 0.f;
      for (int j = 0; j < 16; ++j) t += red[row * 16 + j];
      rowred[row] = t;
    }
    __syncthreads();
    float inv = 1.0f / rowred[row];
    float4* mr = (float4*)(meanL + row * S_DIM + seg * 128);
    for (int j = 0; j < 32; ++j) {
      float4 v = sr[j];
      v.x *= inv; v.y *= inv; v.z *= inv; v.w *= inv;
      sr[j] = v;
      float4 a = mr[j];
      a.x += v.x * (1.0f / H_DIM); a.y += v.y * (1.0f / H_DIM);
      a.z += v.z * (1.0f / H_DIM); a.w += v.w * (1.0f / H_DIM);
      mr[j] = a;
    }
    __syncthreads();

    // ---- ctx = probs x V : all 8 waves (nt = col slice, kh = K half),
    //      4 independent accumulator chains per wave ----
    {
      const int nt = wave & 3, kh = wave >> 2;
      const __bf16* vp = vT + ((size_t)b * E_DIM + ecol + nt * 16 + l16) * S_DIM
                         + kh * (S_DIM / 2) + half * 8;
      const float* pr = scoreL + l16 * S_DIM + kh * (S_DIM / 2) + half * 8;
      v8f c0 = {}, c1 = {}, c2 = {}, c3 = {};
      for (int ks = 0; ks < 8; ++ks) {        // 8 iters x 4 chains = 32 k-steps
        __builtin_prefetch(vp + 128, 0, 0);
        v16bf f0 = frag_from_lds(pr);
        v16bf g0 = load_frag(vp);
        c0 = __builtin_amdgcn_wmma_f32_16x16x32_bf16(false, f0, false, g0, (short)0, c0, false, false);
        v16bf f1 = frag_from_lds(pr + 32);
        v16bf g1 = load_frag(vp + 32);
        c1 = __builtin_amdgcn_wmma_f32_16x16x32_bf16(false, f1, false, g1, (short)0, c1, false, false);
        v16bf f2 = frag_from_lds(pr + 64);
        v16bf g2 = load_frag(vp + 64);
        c2 = __builtin_amdgcn_wmma_f32_16x16x32_bf16(false, f2, false, g2, (short)0, c2, false, false);
        v16bf f3 = frag_from_lds(pr + 96);
        v16bf g3 = load_frag(vp + 96);
        c3 = __builtin_amdgcn_wmma_f32_16x16x32_bf16(false, f3, false, g3, (short)0, c3, false, false);
        pr += 128; vp += 128;
      }
      v8f cs = (c0 + c1) + (c2 + c3);
      if (kh == 1) {
#pragma unroll
        for (int i = 0; i < 8; ++i) ctxred[(nt * 32 + lane) * 8 + i] = cs[i];
      }
      __syncthreads();
      if (kh == 0) {
        const int cc = ecol + nt * 16 + l16;
#pragma unroll
        for (int i = 0; i < 8; ++i) {
          float v = cs[i] + ctxred[(nt * 32 + lane) * 8 + i];
          ctx[(size_t)(q0 + i + half * 8) * E_DIM + cc] = (__bf16)v;
        }
      }
    }
    __syncthreads();
  }

  // ---- write head-mean attention weights ----
  float4* ao = (float4*)(attn_out + (size_t)(b * S_DIM + qt * 16) * S_DIM);
  const float4* m4 = (const float4*)meanL;
  for (int i = tid; i < 16 * S_DIM / 4; i += 256) ao[i] = m4[i];
}

// ---------------------------------------------------------------------------
// Final full-row (D=1024) RMSNorm with routed attn_norm_w -> d_out
// ---------------------------------------------------------------------------
__global__ __launch_bounds__(256) void out_rms(
    const float* __restrict__ o_raw, const int* __restrict__ mod_ids,
    const float* __restrict__ anw, float* __restrict__ out)
{
  __shared__ float red[256];
  const int r = blockIdx.x, tid = threadIdx.x;
  float4 v = ((const float4*)(o_raw + (size_t)r * D_DIM))[tid];
  red[tid] = v.x * v.x + v.y * v.y + v.z * v.z + v.w * v.w;
  __syncthreads();
  for (int s = 128; s > 0; s >>= 1) {
    if (tid < s) red[tid] += red[tid + s];
    __syncthreads();
  }
  float sc = rsqrtf(red[0] * (1.0f / D_DIM) + EPS_F);
  int m = mod_ids[r];
  float4 w = ((const float4*)(anw + (size_t)m * D_DIM))[tid];
  float4 o;
  o.x = v.x * sc * w.x; o.y = v.y * sc * w.y;
  o.z = v.z * sc * w.z; o.w = v.w * sc * w.w;
  ((float4*)(out + (size_t)r * D_DIM))[tid] = o;
}

// ---------------------------------------------------------------------------
extern "C" void kernel_launch(void* const* d_in, const int* in_sizes, int n_in,
                              void* d_out, int out_size, void* d_ws, size_t ws_size,
                              hipStream_t stream) {
  const float* x   = (const float*)d_in[0];
  const float* msk = (const float*)d_in[1];
  const int*   mod = (const int*)d_in[2];
  const float* Wq  = (const float*)d_in[3];
  const float* Wk  = (const float*)d_in[4];
  const float* Wv  = (const float*)d_in[5];
  const float* Wo  = (const float*)d_in[6];
  const float* qn  = (const float*)d_in[7];
  const float* kn  = (const float*)d_in[8];
  const float* anw = (const float*)d_in[9];
  float* out = (float*)d_out;

  char* ws = (char*)d_ws;
  __bf16* xb  = (__bf16*)ws; ws += (size_t)ROWS * D_DIM * 2;       //  8 MB
  __bf16* WqT = (__bf16*)ws; ws += (size_t)2 * D_DIM * E_DIM * 2;  //  4 MB
  __bf16* WkT = (__bf16*)ws; ws += (size_t)2 * D_DIM * E_DIM * 2;
  __bf16* WvT = (__bf16*)ws; ws += (size_t)2 * D_DIM * E_DIM * 2;
  __bf16* WoT = (__bf16*)ws; ws += (size_t)2 * E_DIM * D_DIM * 2;
  __bf16* qb  = (__bf16*)ws; ws += (size_t)ROWS * E_DIM * 2;       //  8 MB
  __bf16* kb  = (__bf16*)ws; ws += (size_t)ROWS * E_DIM * 2;
  __bf16* vT  = (__bf16*)ws; ws += (size_t)B_DIM * E_DIM * S_DIM * 2;
  __bf16* ctx = (__bf16*)ws; ws += (size_t)ROWS * E_DIM * 2;
  float*  orw = (float*)ws;  ws += (size_t)ROWS * D_DIM * 4;       // 16 MB

  // precision conversion / weight transposition
  int xn4 = ROWS * D_DIM / 4;
  cvt_f32_bf16<<<(xn4 + 255) / 256, 256, 0, stream>>>(x, xb, xn4);
  int wtot = 2 * D_DIM * E_DIM;
  transpose_cvt<<<(wtot + 255) / 256, 256, 0, stream>>>(Wq, WqT, D_DIM, E_DIM);
  transpose_cvt<<<(wtot + 255) / 256, 256, 0, stream>>>(Wk, WkT, D_DIM, E_DIM);
  transpose_cvt<<<(wtot + 255) / 256, 256, 0, stream>>>(Wv, WvT, D_DIM, E_DIM);
  transpose_cvt<<<(wtot + 255) / 256, 256, 0, stream>>>(Wo, WoT, E_DIM, D_DIM);

  // QKV projections (routed, q/k with fused per-head RMSNorm, v transposed)
  dim3 g(ROWS / 128, E_DIM / 64, 2);
  gemm_routed<0><<<g, 256, 0, stream>>>(xb, WqT, qb, mod, qn);
  gemm_routed<0><<<g, 256, 0, stream>>>(xb, WkT, kb, mod, kn);
  gemm_routed<1><<<g, 256, 0, stream>>>(xb, WvT, vT, mod, nullptr);

  // attention + head-mean weights (256 KB dynamic LDS per workgroup)
  attn_kernel<<<B_DIM * (S_DIM / 16), 256, 2 * 16 * S_DIM * sizeof(float), stream>>>(
      qb, kb, vT, ctx, msk, out + (size_t)ROWS * D_DIM);

  // output projection (routed) then full-row RMSNorm into d_out
  dim3 g2(ROWS / 128, D_DIM / 64, 2);
  gemm_routed<2><<<g2, 256, 0, stream>>>(ctx, WoT, orw, mod, nullptr);
  out_rms<<<ROWS, 256, 0, stream>>>(orw, mod, anw, out);
}
